// MultiHeadAttention_27986006901260
// MI455X (gfx1250) — matrix operand order repaired and checked by reading, not compiled, not run on hardware
//
#include <hip/hip_runtime.h>

// ---------------------------------------------------------------------------
// MHA for MI455X (gfx1250): fp32 -> bf16 WMMA (f32 accum) pipeline.
//   1) gemm_bias_kernel<float,bf16> x3 : Q/K/V projections into ws (bf16)
//   2) flash_attn_kernel             : online-softmax attention, ctx bf16
//   3) gemm_bias_kernel<bf16,float>  : output projection -> d_out (f32)
// Compute-bound (~412 GFLOP vs ~15us of streamed HBM traffic) -> everything
// runs on v_wmma_f32_16x16x32_bf16, with double-buffered LDS staging and
// global_prefetch_b8 to hide latency.
// ---------------------------------------------------------------------------

typedef __bf16 bf16;
typedef __attribute__((ext_vector_type(4)))  float v4f;
typedef __attribute__((ext_vector_type(8)))  float v8f;
typedef __attribute__((ext_vector_type(4)))  bf16  v4bf;
typedef __attribute__((ext_vector_type(8)))  bf16  v8bf;
typedef __attribute__((ext_vector_type(16))) bf16  v16bf;

#define E_DIM  2048
#define HEADS  16
#define NBATCH 4
#define SEQ    2048
#define DHEAD  128

__device__ __forceinline__ v16bf cat16(v8bf lo, v8bf hi) {
  return __builtin_shufflevector(lo, hi, 0,1,2,3,4,5,6,7,8,9,10,11,12,13,14,15);
}

__device__ __forceinline__ v8f zero8() {
  v8f z = {0.f,0.f,0.f,0.f,0.f,0.f,0.f,0.f};
  return z;
}

// Stage 16 contiguous elements into LDS as bf16 (fused f32->bf16 convert).
__device__ __forceinline__ void stage16(const float* __restrict__ g, bf16* l) {
  v4f a = *(const v4f*)(g);
  v4f b = *(const v4f*)(g + 4);
  v4f c = *(const v4f*)(g + 8);
  v4f d = *(const v4f*)(g + 12);
  v8bf y0, y1;
  y0[0]=(bf16)a[0]; y0[1]=(bf16)a[1]; y0[2]=(bf16)a[2]; y0[3]=(bf16)a[3];
  y0[4]=(bf16)b[0]; y0[5]=(bf16)b[1]; y0[6]=(bf16)b[2]; y0[7]=(bf16)b[3];
  y1[0]=(bf16)c[0]; y1[1]=(bf16)c[1]; y1[2]=(bf16)c[2]; y1[3]=(bf16)c[3];
  y1[4]=(bf16)d[0]; y1[5]=(bf16)d[1]; y1[6]=(bf16)d[2]; y1[7]=(bf16)d[3];
  *(v8bf*)(l)     = y0;
  *(v8bf*)(l + 8) = y1;
}
__device__ __forceinline__ void stage16(const bf16* __restrict__ g, bf16* l) {
  *(v8bf*)(l)     = *(const v8bf*)(g);
  *(v8bf*)(l + 8) = *(const v8bf*)(g + 8);
}

// ---------------------------------------------------------------------------
// C(M,N) = A(M,K) * W(N,K)^T + bias(N).  Block tile 128x128, K-step 32,
// double-buffered LDS (one barrier per K-step).  8 waves: wave=(wm 0..3)x
// (wn 0..1); each wave -> 32x64 (2x4 WMMA tiles).
// ---------------------------------------------------------------------------
template <typename AT, typename OT>
__global__ __launch_bounds__(256)
void gemm_bias_kernel(const AT* __restrict__ A, const float* __restrict__ W,
                      const float* __restrict__ bias, OT* __restrict__ C,
                      int M, int N, int K) {
  __shared__ bf16 Al[2][128][40];   // +8 pad: 80B row stride, 16B aligned
  __shared__ bf16 Wl[2][128][40];

  const int tid  = threadIdx.x;
  const int lane = tid & 31;
  const int wv   = tid >> 5;
  const int wm   = wv & 3;
  const int wn   = wv >> 2;
  const int hiH  = (lane >= 16) ? 1 : 0;
  const int l16  = lane & 15;
  const int m0   = blockIdx.y * 128;
  const int n0   = blockIdx.x * 128;

  v8f acc[2][4];
#pragma unroll
  for (int i = 0; i < 2; ++i)
#pragma unroll
    for (int j = 0; j < 4; ++j) acc[i][j] = zero8();

  const int srow = tid >> 1;         // 0..127
  const int scol = (tid & 1) * 16;   // 0 or 16
  const AT*    aSrc = A + (size_t)(m0 + srow) * K + scol;
  const float* wSrc = W + (size_t)(n0 + srow) * K + scol;

  stage16(aSrc, &Al[0][srow][scol]);
  stage16(wSrc, &Wl[0][srow][scol]);
  __syncthreads();

  int buf = 0;
  for (int k0 = 0; k0 < K; k0 += 32) {
    if (k0 + 32 < K) {                       // stage next step into other buf
      stage16(aSrc + (k0 + 32), &Al[buf ^ 1][srow][scol]);
      stage16(wSrc + (k0 + 32), &Wl[buf ^ 1][srow][scol]);
    }
    if (k0 + 64 < K) {                       // pull k+64 toward L2/L0
      __builtin_prefetch(aSrc + (k0 + 64), 0, 1);
      __builtin_prefetch(wSrc + (k0 + 64), 0, 1);
    }

    v16bf afr[2], bfr[4];
#pragma unroll
    for (int i = 0; i < 2; ++i) {
      const int row = wm * 32 + i * 16 + l16;        // A: lane holds row M
      const int c   = hiH * 8;                       // K halves split by lane-half
      afr[i] = cat16(*(const v8bf*)&Al[buf][row][c],
                     *(const v8bf*)&Al[buf][row][c + 16]);
    }
#pragma unroll
    for (int j = 0; j < 4; ++j) {
      const int row = wn * 64 + j * 16 + l16;        // B: lane holds col N
      const int c   = hiH * 8;                       // K halves split by lane-half
      bfr[j] = cat16(*(const v8bf*)&Wl[buf][row][c],
                     *(const v8bf*)&Wl[buf][row][c + 16]);
    }
#pragma unroll
    for (int i = 0; i < 2; ++i)
#pragma unroll
      for (int j = 0; j < 4; ++j)
        acc[i][j] = __builtin_amdgcn_wmma_f32_16x16x32_bf16(
            false, afr[i], false, bfr[j], (short)0, acc[i][j], false, false);
    __syncthreads();
    buf ^= 1;
  }

#pragma unroll
  for (int j = 0; j < 4; ++j) {
    const int gn = n0 + wn * 64 + j * 16 + l16;
    const float bv = bias[gn];
#pragma unroll
    for (int i = 0; i < 2; ++i) {
#pragma unroll
      for (int r = 0; r < 8; ++r) {
        const int gm = m0 + wm * 32 + i * 16 + r + hiH * 8;
        C[(size_t)gm * N + gn] = (OT)(acc[i][j][r] + bv);
      }
    }
  }
}

// ---------------------------------------------------------------------------
// Flash attention: block = (n,h, 128 query rows); wave owns 16 query rows.
// Double-buffered K/V staging; per 64-key chunk: QK^T (16 wmma) -> mask/scale
// -> online softmax (shfl_xor row reductions within 16-lane halves) -> P via
// per-wave LDS tile -> P*V (16 wmma).
// ---------------------------------------------------------------------------
__global__ __launch_bounds__(256)
void flash_attn_kernel(const bf16* __restrict__ Q, const bf16* __restrict__ Kb,
                       const bf16* __restrict__ Vb, const int* __restrict__ mask,
                       bf16* __restrict__ ctx) {
  __shared__ bf16 Kl[2][64][136];   // 64 keys x 128 d (+8 pad)
  __shared__ bf16 Vt[2][128][72];   // V transposed: d x keys (+8 pad)
  __shared__ bf16 Pl[8][16][72];    // per-wave P tile 16x64 (+8 pad)

  const int tid  = threadIdx.x;
  const int lane = tid & 31;
  const int wv   = tid >> 5;
  const int hiH  = (lane >= 16) ? 1 : 0;
  const int l16  = lane & 15;

  const int b  = blockIdx.y / HEADS;
  const int h  = blockIdx.y % HEADS;
  const int s0 = blockIdx.x * 128;
  const int sRow0 = s0 + wv * 16;

  // staging geometry
  const int kRow = tid >> 2;               // 0..63 (K rows, 32-d slices)
  const int kCol = (tid & 3) * 32;
  const int vRg  = tid >> 4;               // 0..15 -> 4 key rows each
  const int vCg  = (tid & 15) * 8;         // 8 d columns
  const bf16* kSrc = Kb + ((size_t)b * SEQ + kRow) * E_DIM + h * DHEAD + kCol;
  const bf16* vSrc = Vb + ((size_t)b * SEQ + vRg * 4) * E_DIM + h * DHEAD + vCg;

  // Stage one 64-key chunk: K row-major, V transposed with packed b64 stores.
  auto stageKV = [&](int p, int t0) {
    const bf16* kr = kSrc + (size_t)t0 * E_DIM;
#pragma unroll
    for (int i = 0; i < 4; ++i)
      *(v8bf*)&Kl[p][kRow][kCol + i * 8] = *(const v8bf*)(kr + i * 8);

    v8bf x[4];
#pragma unroll
    for (int i = 0; i < 4; ++i)
      x[i] = *(const v8bf*)(vSrc + (size_t)(t0 + i) * E_DIM);
#pragma unroll
    for (int e = 0; e < 8; ++e) {
      v4bf p4;
      p4[0] = x[0][e]; p4[1] = x[1][e]; p4[2] = x[2][e]; p4[3] = x[3][e];
      *(v4bf*)&Vt[p][vCg + e][vRg * 4] = p4;   // 8B packed transpose store
    }
  };

  // Q fragments resident in registers (16 rows x 128 d, A-layout from global)
  v16bf qf[4];
  {
    const bf16* qrow = Q + ((size_t)b * SEQ + (sRow0 + l16)) * E_DIM + h * DHEAD;
#pragma unroll
    for (int kk = 0; kk < 4; ++kk) {
      const int c = kk * 32 + hiH * 8;
      qf[kk] = cat16(*(const v8bf*)(qrow + c), *(const v8bf*)(qrow + c + 16));
    }
  }

  v8f o[8];
#pragma unroll
  for (int j = 0; j < 8; ++j) o[j] = zero8();
  float rmax[8], rsum[8];
#pragma unroll
  for (int r = 0; r < 8; ++r) { rmax[r] = -1e30f; rsum[r] = 0.0f; }

  const float scale = 0.08838834764831845f;  // 1/sqrt(128)

  stageKV(0, 0);
  __syncthreads();

  int buf = 0;
  for (int t0 = 0; t0 < SEQ; t0 += 64) {
    if (t0 + 64 < SEQ) stageKV(buf ^ 1, t0 + 64);
    if (t0 + 128 < SEQ) {
      __builtin_prefetch(kSrc + (size_t)(t0 + 128) * E_DIM, 0, 1);
      __builtin_prefetch(vSrc + (size_t)(t0 + 128) * E_DIM, 0, 1);
    }

    // ---- scores: 16 rows x 64 keys, K = 128
    v8f sc[4];
#pragma unroll
    for (int j = 0; j < 4; ++j) sc[j] = zero8();
#pragma unroll
    for (int kk = 0; kk < 4; ++kk) {
#pragma unroll
      for (int j = 0; j < 4; ++j) {
        const int trow = j * 16 + l16;
        const int c    = kk * 32 + hiH * 8;
        v16bf bfr = cat16(*(const v8bf*)&Kl[buf][trow][c],
                          *(const v8bf*)&Kl[buf][trow][c + 16]);
        sc[j] = __builtin_amdgcn_wmma_f32_16x16x32_bf16(
            false, qf[kk], false, bfr, (short)0, sc[j], false, false);
      }
    }

    // ---- mask + scale (mask[t,s] == 0 -> -inf before /sqrt(d))
#pragma unroll
    for (int j = 0; j < 4; ++j) {
      const int tg = t0 + j * 16 + l16;
#pragma unroll
      for (int r = 0; r < 8; ++r) {
        const int sg = sRow0 + r + hiH * 8;
        const int mk = mask[(size_t)tg * SEQ + sg];
        float v = sc[j][r];
        v = (mk == 0) ? -1e30f : v;
        sc[j][r] = v * scale;
      }
    }

    // ---- online softmax (row stats across 16 lanes of each half-wave)
    float nmax[8], corr[8], lsum[8];
#pragma unroll
    for (int r = 0; r < 8; ++r) {
      float m = fmaxf(fmaxf(sc[0][r], sc[1][r]), fmaxf(sc[2][r], sc[3][r]));
#pragma unroll
      for (int xm = 1; xm < 16; xm <<= 1) m = fmaxf(m, __shfl_xor(m, xm, 32));
      nmax[r] = fmaxf(rmax[r], m);
      corr[r] = __expf(rmax[r] - nmax[r]);
      rmax[r] = nmax[r];
    }
#pragma unroll
    for (int j = 0; j < 8; ++j)
#pragma unroll
      for (int r = 0; r < 8; ++r) o[j][r] *= corr[r];

#pragma unroll
    for (int r = 0; r < 8; ++r) lsum[r] = 0.0f;
#pragma unroll
    for (int j = 0; j < 4; ++j) {
#pragma unroll
      for (int r = 0; r < 8; ++r) {
        const float p = __expf(sc[j][r] - nmax[r]);
        lsum[r] += p;
        Pl[wv][r + hiH * 8][j * 16 + l16] = (bf16)p;   // C-layout -> LDS
      }
    }
#pragma unroll
    for (int r = 0; r < 8; ++r) {
      float s = lsum[r];
#pragma unroll
      for (int xm = 1; xm < 16; xm <<= 1) s += __shfl_xor(s, xm, 32);
      rsum[r] = rsum[r] * corr[r] + s;
    }
    __syncthreads();   // P tiles visible (and next-chunk staging done)

    // ---- O += P @ V   (M=16, K=64 keys, N=128 d)
#pragma unroll
    for (int kk = 0; kk < 2; ++kk) {
      const int ca = kk * 32 + hiH * 8;
      v16bf afr = cat16(*(const v8bf*)&Pl[wv][l16][ca],
                        *(const v8bf*)&Pl[wv][l16][ca + 16]);
#pragma unroll
      for (int j = 0; j < 8; ++j) {
        const int e  = j * 16 + l16;
        const int cb = kk * 32 + hiH * 8;
        v16bf bfr = cat16(*(const v8bf*)&Vt[buf][e][cb],
                          *(const v8bf*)&Vt[buf][e][cb + 16]);
        o[j] = __builtin_amdgcn_wmma_f32_16x16x32_bf16(
            false, afr, false, bfr, (short)0, o[j], false, false);
      }
    }
    __syncthreads();   // K/V/P reads done -> next iter may overwrite
    buf ^= 1;
  }

  // ---- normalize + store ctx (bf16)
#pragma unroll
  for (int r = 0; r < 8; ++r) rsum[r] = 1.0f / rsum[r];
#pragma unroll
  for (int j = 0; j < 8; ++j) {
    const int e = h * DHEAD + j * 16 + l16;
#pragma unroll
    for (int r = 0; r < 8; ++r) {
      const int sg = sRow0 + r + hiH * 8;
      ctx[((size_t)b * SEQ + sg) * E_DIM + e] = (bf16)(o[j][r] * rsum[r]);
    }
  }
}

// ---------------------------------------------------------------------------
extern "C" void kernel_launch(void* const* d_in, const int* in_sizes, int n_in,
                              void* d_out, int out_size, void* d_ws, size_t ws_size,
                              hipStream_t stream) {
  (void)in_sizes; (void)n_in; (void)out_size; (void)ws_size;

  const float* query = (const float*)d_in[0];
  const float* key_  = (const float*)d_in[1];
  const float* value = (const float*)d_in[2];
  const int*   amask = (const int*)d_in[3];
  const float* Wq = (const float*)d_in[4];
  const float* bq = (const float*)d_in[5];
  const float* Wk = (const float*)d_in[6];
  const float* bk = (const float*)d_in[7];
  const float* Wv = (const float*)d_in[8];
  const float* bv = (const float*)d_in[9];
  const float* Wp = (const float*)d_in[10];
  const float* bp = (const float*)d_in[11];
  float* out = (float*)d_out;

  const size_t tokens = (size_t)NBATCH * SEQ;   // 8192
  const size_t elems  = tokens * E_DIM;         // 16.8M
  bf16* Qb  = (bf16*)d_ws;                      // ws layout: 4 x 32 MiB bf16
  bf16* Kb  = Qb + elems;
  bf16* Vb  = Kb + elems;
  bf16* Ctx = Vb + elems;

  dim3 gg(E_DIM / 128, (unsigned)(tokens / 128));
  gemm_bias_kernel<float, bf16><<<gg, 256, 0, stream>>>(query, Wq, bq, Qb,
                                                        (int)tokens, E_DIM, E_DIM);
  gemm_bias_kernel<float, bf16><<<gg, 256, 0, stream>>>(key_,  Wk, bk, Kb,
                                                        (int)tokens, E_DIM, E_DIM);
  gemm_bias_kernel<float, bf16><<<gg, 256, 0, stream>>>(value, Wv, bv, Vb,
                                                        (int)tokens, E_DIM, E_DIM);

  dim3 gf(SEQ / 128, NBATCH * HEADS);
  flash_attn_kernel<<<gf, 256, 0, stream>>>(Qb, Kb, Vb, amask, Ctx);

  gemm_bias_kernel<bf16, float><<<gg, 256, 0, stream>>>(Ctx, Wp, bp, out,
                                                        (int)tokens, E_DIM, E_DIM);
}
